// EntityAttentionDecoder_44856638439565
// MI455X (gfx1250) — matrix operand order repaired and checked by reading, not compiled, not run on hardware
//
#include <hip/hip_runtime.h>
#include <hip/hip_bf16.h>
#include <math.h>

// ---------------------------------------------------------------------------
// EntityAttentionDecoder for MI455X (gfx1250, wave32, WMMA)
//  L=9, E=1024, H=16, Dh=64, F=4096, C=2, B=8, SA=SB=512
// GEMMs: v_wmma_f32_16x16x32_f16, 32x64 per wave (8 WMMA / k-step),
// A-tile staged in LDS via global_load_async_to_lds_b128 (ASYNCcnt).
// Attention: flash-style, one (b,h,q16) tile per wave.
// ---------------------------------------------------------------------------

typedef __attribute__((ext_vector_type(16))) _Float16 v16h;
typedef __attribute__((ext_vector_type(8)))  float    v8f;

#define E_DIM 1024
#define F_DIM 4096
#define H_DIM 16
#define DH    64
#define SA_   512
#define SB_   512
#define B_    8
#define L_    9
#define M_ROWS 4096         // B_*SA_ == B_*SB_
#define EPS_  1e-5f
#define WEIGHT_ 5.0f

__device__ __forceinline__ v8f wmma_f16(v16h a, v16h b, v8f c) {
  // D = A(16x32 f16) * B(32x16 f16) + C(16x16 f32)
  return __builtin_amdgcn_wmma_f32_16x16x32_f16(
      /*neg_a=*/false, a, /*neg_b=*/false, b,
      /*c_mod=*/(short)0, c, /*reuse_a=*/false, /*reuse_b=*/false);
}

// LDS byte-offset of a generic pointer that points into shared memory.
__device__ __forceinline__ uint32_t lds_off_of(const void* p) {
  return (uint32_t)(uintptr_t)(__attribute__((address_space(3))) const void*)p;
}

__device__ __forceinline__ void wait_asynccnt0() {
#if __has_builtin(__builtin_amdgcn_s_wait_asynccnt)
  __builtin_amdgcn_s_wait_asynccnt(0);
#else
  asm volatile("s_wait_asynccnt 0x0" ::: "memory");
#endif
}

// A fragment (16x32, M x K), f16, from row-major f32 matrix.
// Contiguous k order used here (not AMD's table order): lane L holds row m=L&15; K = kbase + h,
// kbase = (L>>4)*16.  16 contiguous elements per lane.
__device__ __forceinline__ v16h load_a_frag_f32(const float* __restrict__ A,
                                                int lda, int row0, int k0,
                                                int lane) {
  int m  = lane & 15;
  int kb = (lane >> 4) * 16;
  const float* p = A + (size_t)(row0 + m) * lda + k0 + kb;
  v16h a;
#pragma unroll
  for (int i = 0; i < 8; ++i) {
    a[i]     = (_Float16)p[i];
    a[i + 8] = (_Float16)p[i + 8];
  }
  return a;
}

// B fragment (32x16, K x N), f16, sourced from row-major f32 "weight" matrix
// W[N][K] (so B[k][n] = W[n][k]).  Contiguous k order (as for A, not AMD's table order): lane L holds column
// n = L&15, K = kbase2 + h, kbase2 = (L>>4)*16 -> 16 contiguous f32 per lane.
__device__ __forceinline__ v16h load_b_frag_f32(const float* __restrict__ W,
                                                int ldw, int col0, int k0,
                                                int lane) {
  int n  = lane & 15;
  int kb = (lane >> 4) * 16;
  const float* p = W + (size_t)(col0 + n) * ldw + k0 + kb;
  v16h b;
#pragma unroll
  for (int i = 0; i < 16; ++i) b[i] = (_Float16)p[i];
  return b;
}

// ---------------------------------------------------------------------------
// GEMM:  C[M,N] = act( A[M,K] @ W[N,K]^T + bias[N] )    (einsum 'me,ne->mn')
// Block: 256 threads = 8 waves, arranged 2 (rows) x 4 (cols).
// Block tile: 64 rows x 256 cols.  Per wave: 32 rows x 64 cols
// (2 m-frags x 4 n-frags = 8 accumulators, 8 WMMA per 32-deep k-step).
// The 64x32 f32 A-tile is async-copied to LDS once per k-step and shared
// by all column wave-groups.
// ---------------------------------------------------------------------------
__global__ __launch_bounds__(256) void gemm_wmma_f32(
    float* __restrict__ C, const float* __restrict__ A,
    const float* __restrict__ W, const float* __restrict__ bias,
    int M, int N, int K, int relu) {
  __shared__ float sA[64 * 32];            // 8 KB A staging tile

  int tid  = threadIdx.x;
  int lane = tid & 31;
  int wave = tid >> 5;
  int wrow = wave >> 2;                    // 0..1
  int wcol = wave & 3;                     // 0..3
  int row0b = blockIdx.y * 64;             // block row origin
  int col0  = blockIdx.x * 256 + wcol * 64;

  // staging assignment: 2048 floats, 8 per thread (two b128 per lane)
  int f  = tid * 8;
  int sr = f >> 5;                         // 0..63
  int sc = f & 31;                         // 0,8,16,24

  v8f acc[2][4] = {};

  for (int k0 = 0; k0 < K; k0 += 32) {
    // ---- async stage A tile (64 x 32 f32) into LDS ----------------------
    {
      const float* ga = A + (size_t)(row0b + sr) * K + k0 + sc;
      uint32_t l0 = lds_off_of(&sA[sr * 32 + sc]);
      asm volatile("global_load_async_to_lds_b128 %0, %1, off"
                   :: "v"(l0), "v"(ga) : "memory");
      asm volatile("global_load_async_to_lds_b128 %0, %1, off"
                   :: "v"(l0 + 16u), "v"(ga + 4) : "memory");
    }
    // prefetch next weight k-tile while async copy is in flight
    if (k0 + 32 < K)
      __builtin_prefetch(W + (size_t)(col0 + (lane & 15)) * K + k0 + 32, 0, 1);
    wait_asynccnt0();
    __syncthreads();

    // ---- A fragments from LDS (shared across column wave-groups) -------
    v16h afrag[2];
#pragma unroll
    for (int mt = 0; mt < 2; ++mt) {
      int mloc = wrow * 32 + mt * 16 + (lane & 15);
      int kb   = (lane >> 4) * 16;
      const float* p = &sA[mloc * 32 + kb];
      v16h a;
#pragma unroll
      for (int i = 0; i < 8; ++i) {
        a[i]     = (_Float16)p[i];
        a[i + 8] = (_Float16)p[i + 8];
      }
      afrag[mt] = a;
    }

    // ---- B fragments from global, 8 WMMAs -------------------------------
#pragma unroll
    for (int nt = 0; nt < 4; ++nt) {
      v16h bf = load_b_frag_f32(W, K, col0 + nt * 16, k0, lane);
      acc[0][nt] = wmma_f16(afrag[0], bf, acc[0][nt]);
      acc[1][nt] = wmma_f16(afrag[1], bf, acc[1][nt]);
    }
    __syncthreads();                       // protect sA before next stage
  }

  // ---- epilogue: bias (+relu) and store ---------------------------------
  int n  = lane & 15;
  int mh = (lane >> 4) * 8;                // C layout: VGPR r -> row mh+r
#pragma unroll
  for (int nt = 0; nt < 4; ++nt) {
    float bv = bias[col0 + nt * 16 + n];
#pragma unroll
    for (int mt = 0; mt < 2; ++mt) {
#pragma unroll
      for (int r = 0; r < 8; ++r) {
        float v = acc[mt][nt][r] + bv;
        if (relu) v = fmaxf(v, 0.0f);
        size_t row = (size_t)(row0b + wrow * 32 + mt * 16 + mh + r);
        C[row * N + col0 + nt * 16 + n] = v;
      }
    }
  }
}

// ---------------------------------------------------------------------------
// Flash cross-attention. One wave handles one (b, h, 16-query tile).
// S = (Q Kt) * 1/8, online softmax, O += P V. P is restaged through a
// per-wave LDS tile to convert C-layout -> A-layout for the PV WMMA.
// 4096 wave-tasks total -> grid 1024 x 128 threads (4 waves/block).
// ---------------------------------------------------------------------------
__global__ __launch_bounds__(128) void attn_flash(
    float* __restrict__ ctx, const float* __restrict__ Q,
    const float* __restrict__ Km, const float* __restrict__ Vm) {
  __shared__ _Float16 lds_p[4][16 * 32];   // per-wave P tile (16 rows x 32 keys)

  int lane = threadIdx.x & 31;
  int wave = threadIdx.x >> 5;
  int t  = blockIdx.x * 4 + wave;          // 0..4095
  int qb = t & 31;                         // SA/16
  int h  = (t >> 5) & (H_DIM - 1);
  int b  = t >> 9;
  const float scale = 0.125f;              // 1/sqrt(64)

  const float* Qp = Q  + ((size_t)b * SA_ + qb * 16) * E_DIM + h * DH;
  const float* Kp = Km + (size_t)b * SB_ * E_DIM + h * DH;
  const float* Vp = Vm + (size_t)b * SB_ * E_DIM + h * DH;

  // Q tile 16x64 -> two A fragments (K-dim split 0..31 / 32..63)
  v16h qa0 = load_a_frag_f32(Qp, E_DIM, 0, 0,  lane);
  v16h qa1 = load_a_frag_f32(Qp, E_DIM, 0, 32, lane);

  v8f oacc[4] = {};                        // O: 16 x 64 (4 d-tiles of 16)
  float mrow[8], lrow[8];
#pragma unroll
  for (int r = 0; r < 8; ++r) { mrow[r] = -INFINITY; lrow[r] = 0.0f; }

  _Float16* P = lds_p[wave];
  int cn = lane & 15;
  int cm = (lane >> 4) * 8;

  for (int kc = 0; kc < SB_; kc += 32) {
    // --- S = Q K^T for 32 keys: two 16x16 tiles --------------------------
    const float* Kc = Kp + (size_t)kc * E_DIM;
    v8f s0 = {}, s1 = {};
    {
      v16h kb0 = load_b_frag_f32(Kc, E_DIM, 0, 0,  lane);
      s0 = wmma_f16(qa0, kb0, s0);
      v16h kb1 = load_b_frag_f32(Kc, E_DIM, 0, 32, lane);
      s0 = wmma_f16(qa1, kb1, s0);
    }
    {
      v16h kb0 = load_b_frag_f32(Kc, E_DIM, 16, 0,  lane);
      s1 = wmma_f16(qa0, kb0, s1);
      v16h kb1 = load_b_frag_f32(Kc, E_DIM, 16, 32, lane);
      s1 = wmma_f16(qa1, kb1, s1);
    }

    // --- online softmax (row-wise across 16 lanes of each half) ----------
#pragma unroll
    for (int r = 0; r < 8; ++r) {
      float a0 = s0[r] * scale;
      float a1 = s1[r] * scale;
      float mx = fmaxf(a0, a1);
#pragma unroll
      for (int off = 8; off; off >>= 1) mx = fmaxf(mx, __shfl_xor(mx, off, 32));
      float mnew  = fmaxf(mrow[r], mx);
      float alpha = __expf(mrow[r] - mnew);
      float p0 = __expf(a0 - mnew);
      float p1 = __expf(a1 - mnew);
      float ss = p0 + p1;
#pragma unroll
      for (int off = 8; off; off >>= 1) ss += __shfl_xor(ss, off, 32);
      lrow[r] = lrow[r] * alpha + ss;
      mrow[r] = mnew;
#pragma unroll
      for (int dt = 0; dt < 4; ++dt) oacc[dt][r] *= alpha;
      s0[r] = p0;
      s1[r] = p1;
    }

    // --- restage P: C-layout regs -> row-major 16x32 LDS -----------------
    __syncthreads();
#pragma unroll
    for (int r = 0; r < 8; ++r) {
      P[(cm + r) * 32 + cn]      = (_Float16)s0[r];
      P[(cm + r) * 32 + 16 + cn] = (_Float16)s1[r];
    }
    __syncthreads();

    // A fragment of P (16 rows x 32 keys)
    v16h pa;
    {
      int m  = lane & 15;
      int kb = (lane >> 4) * 16;
#pragma unroll
      for (int i = 0; i < 8; ++i) {
        pa[i]     = P[m * 32 + kb + i];
        pa[i + 8] = P[m * 32 + kb + 8 + i];
      }
    }

    // --- O += P V  (4 d-tiles of width 16) -------------------------------
#pragma unroll
    for (int dt = 0; dt < 4; ++dt) {
      int vn  = lane & 15;
      int kb2 = (lane >> 4) * 16;
      const float* vp = Vp + (size_t)(kc + kb2) * E_DIM + dt * 16 + vn;
      v16h vb;
#pragma unroll
      for (int i = 0; i < 16; ++i) vb[i] = (_Float16)vp[(size_t)i * E_DIM];
      oacc[dt] = wmma_f16(pa, vb, oacc[dt]);
    }
    __syncthreads();
  }

  // --- normalize and store ctx[b, q, h*64 + d] ---------------------------
#pragma unroll
  for (int r = 0; r < 8; ++r) {
    float inv = (lrow[r] > 0.0f) ? 1.0f / lrow[r] : 0.0f;
    size_t row = (size_t)b * SA_ + qb * 16 + cm + r;
#pragma unroll
    for (int dt = 0; dt < 4; ++dt)
      ctx[row * E_DIM + h * DH + dt * 16 + cn] = oacc[dt][r] * inv;
  }
}

// ---------------------------------------------------------------------------
// out[row] = LayerNorm(x[row] + res[row]) * g + b,  E=1024, block=256
// ---------------------------------------------------------------------------
__global__ __launch_bounds__(256) void ln_residual(
    float* __restrict__ out, const float* __restrict__ x,
    const float* __restrict__ res, const float* __restrict__ g,
    const float* __restrict__ bta) {
  __shared__ float red[8];
  int row  = blockIdx.x;
  int tid  = threadIdx.x;
  int lane = tid & 31;
  int wid  = tid >> 5;
  const float* xp = x   + (size_t)row * E_DIM;
  const float* rp = res + (size_t)row * E_DIM;

  float v[4];
  float s = 0.0f;
#pragma unroll
  for (int i = 0; i < 4; ++i) {
    int c = tid + i * 256;
    v[i] = xp[c] + rp[c];
    s += v[i];
  }
#pragma unroll
  for (int off = 16; off; off >>= 1) s += __shfl_xor(s, off, 32);
  if (lane == 0) red[wid] = s;
  __syncthreads();
  float tot = 0.0f;
#pragma unroll
  for (int i = 0; i < 8; ++i) tot += red[i];
  float mean = tot * (1.0f / (float)E_DIM);
  __syncthreads();

  float s2 = 0.0f;
#pragma unroll
  for (int i = 0; i < 4; ++i) {
    float d = v[i] - mean;
    s2 += d * d;
  }
#pragma unroll
  for (int off = 16; off; off >>= 1) s2 += __shfl_xor(s2, off, 32);
  if (lane == 0) red[wid] = s2;
  __syncthreads();
  float tot2 = 0.0f;
#pragma unroll
  for (int i = 0; i < 8; ++i) tot2 += red[i];
  float inv = rsqrtf(tot2 * (1.0f / (float)E_DIM) + EPS_);

#pragma unroll
  for (int i = 0; i < 4; ++i) {
    int c = tid + i * 256;
    out[(size_t)row * E_DIM + c] = (v[i] - mean) * inv * g[c] + bta[c];
  }
}

// ---------------------------------------------------------------------------
// Classifier head + weighted CE loss + predicted labels.
// ---------------------------------------------------------------------------
__global__ void loss_init(float* accum) {
  if (threadIdx.x < 2) accum[threadIdx.x] = 0.0f;
}

__global__ __launch_bounds__(256) void cls_loss(
    const float* __restrict__ Aout, const float* __restrict__ cls_w,
    const float* __restrict__ cls_b, const int* __restrict__ maskA,
    const int* __restrict__ cand, const int* __restrict__ labels,
    float* __restrict__ pred_out, float* __restrict__ accum) {
  int idx = blockIdx.x * blockDim.x + threadIdx.x;
  if (idx >= M_ROWS) return;
  const float* a = Aout + (size_t)idx * E_DIM;
  float e0 = cls_b[0], e1 = cls_b[1];
  for (int i = 0; i < E_DIM; ++i) {
    float av = a[i];
    e0 += av * cls_w[i];
    e1 += av * cls_w[E_DIM + i];
  }
  float mx  = fmaxf(e0, e1);
  float lse = mx + __logf(__expf(e0 - mx) + __expf(e1 - mx));
  int   lab = labels[idx];
  float ce  = lse - ((lab == 1) ? e1 : e0);
  float w   = (lab == 1) ? WEIGHT_ : 1.0f;
  float valid = (maskA[idx] == 1) ? 1.0f : 0.0f;
  atomicAdd(&accum[0], ce * w * valid);
  atomicAdd(&accum[1], valid);
  int predicted = (e1 > e0) ? 1 : 0;       // argmax; tie -> class 0
  pred_out[idx] = (float)((cand[idx] == 1 && predicted == 1) ? 1 : 0);
}

__global__ void loss_final(const float* __restrict__ accum,
                           float* __restrict__ out) {
  if (threadIdx.x == 0) out[0] = accum[0] / fmaxf(accum[1], 1.0f);
}

// ---------------------------------------------------------------------------
extern "C" void kernel_launch(void* const* d_in, const int* in_sizes, int n_in,
                              void* d_out, int out_size, void* d_ws,
                              size_t ws_size, hipStream_t stream) {
  (void)in_sizes; (void)n_in; (void)out_size; (void)ws_size;
  const float* encoded  = (const float*)d_in[0];
  const float* describe = (const float*)d_in[1];
  const int*   maskA    = (const int*)d_in[2];
  const int*   cand     = (const int*)d_in[3];
  const int*   labels   = (const int*)d_in[4];
  const float* in_proj_w = (const float*)d_in[5];
  const float* in_proj_b = (const float*)d_in[6];
  const float* out_w  = (const float*)d_in[7];
  const float* out_b  = (const float*)d_in[8];
  const float* ln1_g  = (const float*)d_in[9];
  const float* ln1_b  = (const float*)d_in[10];
  const float* ffn_w1 = (const float*)d_in[11];
  const float* ffn_b1 = (const float*)d_in[12];
  const float* ffn_w2 = (const float*)d_in[13];
  const float* ffn_b2 = (const float*)d_in[14];
  const float* ln2_g  = (const float*)d_in[15];
  const float* ln2_b  = (const float*)d_in[16];
  const float* cls_w  = (const float*)d_in[17];
  const float* cls_b  = (const float*)d_in[18];
  float* out = (float*)d_out;

  const size_t ME = (size_t)M_ROWS * E_DIM;  // 4,194,304 floats
  float* ws    = (float*)d_ws;
  float* A_cur = ws;
  float* Qb    = ws + 1 * ME;
  float* Kb    = ws + 2 * ME;
  float* Vb    = ws + 3 * ME;
  float* Cx    = ws + 4 * ME;
  float* Tmp   = ws + 5 * ME;   // attn_out / FFN output
  float* A1    = ws + 6 * ME;
  float* Hb    = ws + 7 * ME;   // 4096 x 4096 hidden (4*ME)
  float* accum = ws + 11 * ME;

  hipMemcpyAsync(A_cur, encoded, ME * sizeof(float),
                 hipMemcpyDeviceToDevice, stream);

  dim3 blk(256);
  auto gemm = [&](float* C, const float* A, const float* W, const float* bias,
                  int M, int N, int K, int relu) {
    dim3 grid(N / 256, M / 64);
    gemm_wmma_f32<<<grid, blk, 0, stream>>>(C, A, W, bias, M, N, K, relu);
  };

  for (int l = 0; l < L_; ++l) {
    const float* wq = in_proj_w + (size_t)l * 3 * E_DIM * E_DIM;
    const float* wk = wq + (size_t)E_DIM * E_DIM;
    const float* wv = wq + 2 * (size_t)E_DIM * E_DIM;
    const float* bq = in_proj_b + (size_t)l * 3 * E_DIM;
    const float* bk = bq + E_DIM;
    const float* bv = bq + 2 * E_DIM;
    const float* ow = out_w + (size_t)l * E_DIM * E_DIM;
    const float* ob = out_b + (size_t)l * E_DIM;
    const float* W1 = ffn_w1 + (size_t)l * F_DIM * E_DIM;
    const float* B1 = ffn_b1 + (size_t)l * F_DIM;
    const float* W2 = ffn_w2 + (size_t)l * E_DIM * F_DIM;
    const float* B2 = ffn_b2 + (size_t)l * E_DIM;

    // Q/K/V projections
    gemm(Qb, A_cur,    wq, bq, M_ROWS, E_DIM, E_DIM, 0);
    gemm(Kb, describe, wk, bk, M_ROWS, E_DIM, E_DIM, 0);
    gemm(Vb, describe, wv, bv, M_ROWS, E_DIM, E_DIM, 0);

    // flash cross-attention -> ctx
    attn_flash<<<dim3(1024), dim3(128), 0, stream>>>(Cx, Qb, Kb, Vb);

    // output projection + LN(residual)
    gemm(Tmp, Cx, ow, ob, M_ROWS, E_DIM, E_DIM, 0);
    ln_residual<<<dim3(M_ROWS), blk, 0, stream>>>(
        A1, Tmp, A_cur, ln1_g + (size_t)l * E_DIM, ln1_b + (size_t)l * E_DIM);

    // FFN: relu(A1 W1^T + b1) W2^T + b2, then LN(residual)
    gemm(Hb, A1, W1, B1, M_ROWS, F_DIM, E_DIM, 1);
    gemm(Tmp, Hb, W2, B2, M_ROWS, E_DIM, F_DIM, 0);
    ln_residual<<<dim3(M_ROWS), blk, 0, stream>>>(
        A_cur, Tmp, A1, ln2_g + (size_t)l * E_DIM, ln2_b + (size_t)l * E_DIM);
  }

  // classifier + loss + predictions
  loss_init<<<dim3(1), dim3(32), 0, stream>>>(accum);
  cls_loss<<<dim3(M_ROWS / 256), blk, 0, stream>>>(
      A_cur, cls_w, cls_b, maskA, cand, labels, out + 1, accum);
  loss_final<<<dim3(1), dim3(32), 0, stream>>>(accum, out);
}
